// RCNN_29102698398123
// MI455X (gfx1250) — matrix operand order repaired and checked by reading, not compiled, not run on hardware
//
#include <hip/hip_runtime.h>

// ---------------------------------------------------------------------------
// Types for CDNA5 WMMA (wave32)
// ---------------------------------------------------------------------------
typedef __attribute__((ext_vector_type(16))) _Float16 v16h;
typedef __attribute__((ext_vector_type(8)))  _Float16 v8h;
typedef __attribute__((ext_vector_type(8)))  float    v8f;
typedef _Float16 half_t;

// Model dimensions
#define BATCH 256
#define NUMIN 4096
#define C1 64
#define L1 1024     // after pool4 of 4096
#define C2 128
#define TT 256      // after pool4 of 1024 (sequence length)
#define HH 256      // lstm hidden
#define NG 1024     // 4*H gates

// ---------------------------------------------------------------------------
// Fragment loaders (contiguous k order 16*(lane/16)+s, same for A and B)
// A 16x32 f16, row-major source: lane l<16 -> row l, K {kb..kb+15},
// kb = (lane>>4)*16. Two 16B loads.
// ---------------------------------------------------------------------------
__device__ __forceinline__ v16h load_fragA(const half_t* __restrict__ p, int ld) {
    const unsigned lane = threadIdx.x & 31u;
    const unsigned r  = lane & 15u;
    const unsigned kb = (lane >> 4) * 16u;
    const half_t* q = p + (size_t)r * ld + kb;
    v8h lo = *(const v8h*)q;
    v8h hi = *(const v8h*)(q + 8);
    v16h f;
#pragma unroll
    for (int i = 0; i < 8; ++i) { f[i] = lo[i]; f[i + 8] = hi[i]; }
    return f;
}

// B 32x16 f16, column-major source (column = 16 contiguous K halves * 2):
// lane n<16 -> column n, K {0..15}; lane n+16 -> column n, K {16..31}.
__device__ __forceinline__ v16h load_fragB(const half_t* __restrict__ p, int ld) {
    const unsigned lane = threadIdx.x & 31u;
    const half_t* q = p + (size_t)(lane & 15u) * ld + (lane >> 4) * 16u;
    return *(const v16h*)q;
}

// Guarded B loader: column index (col0 + lane%16) checked against [0, ncols), zero-filled outside
__device__ __forceinline__ v16h load_fragB_guard(const half_t* __restrict__ mat,
                                                 int col0, int ncols, int ld, int k0) {
    const unsigned lane = threadIdx.x & 31u;
    int c = col0 + (int)(lane & 15u);
    v16h z = {};
    if ((unsigned)c < (unsigned)ncols) {
        const half_t* q = mat + (size_t)c * ld + k0 + (lane >> 4) * 16u;
        z = *(const v16h*)q;
    }
    return z;
}

#define WMMA_F16(A, B, C) \
    __builtin_amdgcn_wmma_f32_16x16x32_f16(false, (A), false, (B), (short)0, (C), false, false)

__device__ __forceinline__ float sigf(float x) { return 1.0f / (1.0f + __expf(-x)); }

// ---------------------------------------------------------------------------
// conv1 (1->64, K=9, pad=4) + bias + relu + avgpool4 + BN1 stats
// grid: 4096 = 256 b * 16 l-tiles; block: 256
// ---------------------------------------------------------------------------
__global__ void __launch_bounds__(256) conv1_kernel(
        const float* __restrict__ x, const float* __restrict__ w,
        const float* __restrict__ bias, float* __restrict__ y1,
        float* __restrict__ sum1, float* __restrict__ sq1) {
    __shared__ float xs[264];
    __shared__ float bsum[C1], bsq[C1];
    const int b  = blockIdx.x >> 4;
    const int lt = blockIdx.x & 15;
    const int lp0 = lt << 6;            // pooled base, 64 pooled outputs
    const int L0  = lp0 << 2;           // pre-pool base

    for (int idx = threadIdx.x; idx < 264; idx += 256) {
        int gi = L0 - 4 + idx;
        xs[idx] = (gi >= 0 && gi < NUMIN) ? x[(size_t)b * NUMIN + gi] : 0.0f;
    }
    if (threadIdx.x < C1) { bsum[threadIdx.x] = 0.f; bsq[threadIdx.x] = 0.f; }
    __syncthreads();

    const int c = threadIdx.x & 63;
    const int q = threadIdx.x >> 6;     // 0..3
    float wr[9];
#pragma unroll
    for (int k = 0; k < 9; ++k) wr[k] = w[c * 9 + k];
    const float bv = bias[c];

    float lsum = 0.f, lsq = 0.f;
    for (int ii = 0; ii < 16; ++ii) {
        int lpr = q + (ii << 2);        // 0..63
        float acc = 0.f;
#pragma unroll
        for (int p = 0; p < 4; ++p) {
            float s = bv;
#pragma unroll
            for (int k = 0; k < 9; ++k) s += wr[k] * xs[(lpr << 2) + p + k];
            acc += fmaxf(s, 0.f);
        }
        float v = acc * 0.25f;
        y1[(size_t)b * (C1 * L1) + (size_t)c * L1 + lp0 + lpr] = v;
        lsum += v; lsq += v * v;
    }
    atomicAdd(&bsum[c], lsum);
    atomicAdd(&bsq[c],  lsq);
    __syncthreads();
    if (threadIdx.x < C1) {
        atomicAdd(&sum1[threadIdx.x], bsum[threadIdx.x]);
        atomicAdd(&sq1[threadIdx.x],  bsq[threadIdx.x]);
    }
}

// ---------------------------------------------------------------------------
// BN fold: scale = g*rsqrt(var+eps), shift = beta - mu*scale
// ---------------------------------------------------------------------------
__global__ void bn_prep_kernel(const float* __restrict__ sum, const float* __restrict__ sq,
                               const float* __restrict__ gamma, const float* __restrict__ beta,
                               float* __restrict__ scale, float* __restrict__ shift,
                               int C, float invN) {
    int c = blockIdx.x * blockDim.x + threadIdx.x;
    if (c >= C) return;
    float mu  = sum[c] * invN;
    float var = sq[c] * invN - mu * mu;
    float a   = gamma[c] * rsqrtf(var + 1e-5f);
    scale[c] = a;
    shift[c] = beta[c] - mu * a;
}

// y1 [b][c][l] f32 -> normalized f16 transposed y1t [b][l][c]
__global__ void y1_norm_t_kernel(const float* __restrict__ y1,
                                 const float* __restrict__ scale, const float* __restrict__ shift,
                                 half_t* __restrict__ y1t) {
    int tid = blockIdx.x * 256 + threadIdx.x;          // 256*1024*64
    int c = tid & 63;
    int l = (tid >> 6) & 1023;
    int b = tid >> 16;
    float v = y1[(size_t)b * (C1 * L1) + (size_t)c * L1 + l];
    y1t[tid] = (half_t)(v * scale[c] + shift[c]);
}

// ---------------------------------------------------------------------------
// conv2 as WMMA GEMM: 5 shifted taps, K=64 each.
// A = W2H[kt] (128x64 row-major f16), B = y1t columns (64-deep, K-contiguous).
// Fused bias+relu+pool4 in LDS + BN2 stats. grid 4096 = 256 b * 16 tiles; block 256 (8 waves)
// ---------------------------------------------------------------------------
__global__ void __launch_bounds__(256) conv2_wmma_kernel(
        const half_t* __restrict__ y1t, const half_t* __restrict__ W2H,
        const float* __restrict__ cb, float* __restrict__ y2,
        float* __restrict__ sum2, float* __restrict__ sq2) {
    __shared__ float tile[C2 * 64];
    __shared__ float bsum[C2], bsq[C2];
    const int b  = blockIdx.x >> 4;
    const int lt = blockIdx.x & 15;
    const int l0 = lt << 6;                       // pre-pool tile base (64 wide)
    const int m0 = (threadIdx.x >> 5) << 4;       // wave -> 16 output channels
    const half_t* ybase = y1t + (size_t)b * (L1 * C1);

    v8f acc0 = {}, acc1 = {}, acc2 = {}, acc3 = {};
    for (int kt = 0; kt < 5; ++kt) {
#pragma unroll
        for (int kk = 0; kk < 2; ++kk) {
            const int k0 = kk << 5;
            v16h a  = load_fragA(W2H + (size_t)kt * (C2 * C1) + (size_t)m0 * C1 + k0, C1);
            v16h b0 = load_fragB_guard(ybase, l0 +  0 + kt - 2, L1, C1, k0);
            v16h b1 = load_fragB_guard(ybase, l0 + 16 + kt - 2, L1, C1, k0);
            v16h b2 = load_fragB_guard(ybase, l0 + 32 + kt - 2, L1, C1, k0);
            v16h b3 = load_fragB_guard(ybase, l0 + 48 + kt - 2, L1, C1, k0);
            acc0 = WMMA_F16(a, b0, acc0);
            acc1 = WMMA_F16(a, b1, acc1);
            acc2 = WMMA_F16(a, b2, acc2);
            acc3 = WMMA_F16(a, b3, acc3);
        }
    }

    if (threadIdx.x < C2) { bsum[threadIdx.x] = 0.f; bsq[threadIdx.x] = 0.f; }
    const unsigned lane = threadIdx.x & 31u;
    const int col = lane & 15;
    const int rbase = m0 + (int)((lane >> 4) << 3);
#pragma unroll
    for (int r = 0; r < 8; ++r) {
        int row = rbase + r;
        float bv = cb[row];
        tile[row * 64 + col +  0] = fmaxf(acc0[r] + bv, 0.f);
        tile[row * 64 + col + 16] = fmaxf(acc1[r] + bv, 0.f);
        tile[row * 64 + col + 32] = fmaxf(acc2[r] + bv, 0.f);
        tile[row * 64 + col + 48] = fmaxf(acc3[r] + bv, 0.f);
    }
    __syncthreads();

#pragma unroll
    for (int ii = 0; ii < 8; ++ii) {
        int idx = (int)threadIdx.x + (ii << 8);   // 0..2047 = 128ch * 16 pooled
        int ch = idx >> 4;
        int p  = idx & 15;
        const float* tp = tile + ch * 64 + (p << 2);
        float v = 0.25f * (tp[0] + tp[1] + tp[2] + tp[3]);
        y2[(size_t)b * (C2 * TT) + (size_t)ch * TT + (lt << 4) + p] = v;
        atomicAdd(&bsum[ch], v);
        atomicAdd(&bsq[ch],  v * v);
    }
    __syncthreads();
    if (threadIdx.x < C2) {
        atomicAdd(&sum2[threadIdx.x], bsum[threadIdx.x]);
        atomicAdd(&sq2[threadIdx.x],  bsq[threadIdx.x]);
    }
}

// y2 [b][c][t] f32 -> normalized f16 X [t][b][c]
__global__ void y2_norm_t_kernel(const float* __restrict__ y2,
                                 const float* __restrict__ scale, const float* __restrict__ shift,
                                 half_t* __restrict__ X) {
    int tid = blockIdx.x * 256 + threadIdx.x;     // 256*256*128
    int c = tid & 127;
    int b = (tid >> 7) & 255;
    int t = tid >> 15;
    float v = y2[(size_t)b * (C2 * TT) + (size_t)c * TT + t];
    X[tid] = (half_t)(v * scale[c] + shift[c]);
}

// ---------------------------------------------------------------------------
// Weight prep: column-major (K-contiguous) f16 weights, concatenated wi|wh
// ---------------------------------------------------------------------------
__global__ void prep_wcat_kernel(const float* __restrict__ Wa, int Ka,
                                 const float* __restrict__ Wb, int Kb,
                                 half_t* __restrict__ out) {
    int tid = blockIdx.x * 256 + threadIdx.x;
    int K = Ka + Kb;
    if (tid >= NG * K) return;
    int j = tid / K;
    int k = tid - j * K;
    float v = (k < Ka) ? Wa[(size_t)k * NG + j] : Wb[(size_t)(k - Ka) * NG + j];
    out[tid] = (half_t)v;
}

__global__ void prep_w2_kernel(const float* __restrict__ w, half_t* __restrict__ out) {
    int tid = blockIdx.x * 256 + threadIdx.x;     // 5*128*64
    if (tid >= 5 * C2 * C1) return;
    int kt = tid / (C2 * C1);
    int r  = tid - kt * (C2 * C1);
    int o  = r >> 6;
    int i  = r & 63;
    out[tid] = (half_t)w[o * (C1 * 5) + i * 5 + kt];
}

__global__ void prep_bias_kernel(const float* bi0, const float* bh0,
                                 const float* bi1, const float* bh1,
                                 float* bias0, float* bias1) {
    int j = blockIdx.x * 256 + threadIdx.x;
    if (j < NG) { bias0[j] = bi0[j] + bh0[j]; bias1[j] = bi1[j] + bh1[j]; }
}

// ---------------------------------------------------------------------------
// LSTM gate GEMM: G[256 x 1024] = [A1 | A2](f16) @ WT^T, WT col-major [1024][Ktot]
// grid 32 blocks x 256 threads = 256 waves; wave -> 16(M) x 64(N) strip
// ---------------------------------------------------------------------------
__global__ void __launch_bounds__(256) gates_gemm_kernel(
        const half_t* __restrict__ A1, int K1, int lda1,
        const half_t* __restrict__ A2, int lda2,
        int Ktot, const half_t* __restrict__ WT, int ldw,
        float* __restrict__ G) {
    const int wave = blockIdx.x * 8 + (threadIdx.x >> 5);
    const int m0 = (wave >> 4) << 4;
    const int n0 = (wave & 15) << 6;
    v8f acc0 = {}, acc1 = {}, acc2 = {}, acc3 = {};
    for (int k0 = 0; k0 < Ktot; k0 += 32) {
        v16h a;
        if (k0 < K1) a = load_fragA(A1 + (size_t)m0 * lda1 + k0, lda1);
        else         a = load_fragA(A2 + (size_t)m0 * lda2 + (k0 - K1), lda2);
        v16h b0 = load_fragB(WT + (size_t)(n0 +  0) * ldw + k0, ldw);
        v16h b1 = load_fragB(WT + (size_t)(n0 + 16) * ldw + k0, ldw);
        v16h b2 = load_fragB(WT + (size_t)(n0 + 32) * ldw + k0, ldw);
        v16h b3 = load_fragB(WT + (size_t)(n0 + 48) * ldw + k0, ldw);
        acc0 = WMMA_F16(a, b0, acc0);
        acc1 = WMMA_F16(a, b1, acc1);
        acc2 = WMMA_F16(a, b2, acc2);
        acc3 = WMMA_F16(a, b3, acc3);
    }
    const unsigned lane = threadIdx.x & 31u;
    const int col = lane & 15;
    const int rbase = m0 + (int)((lane >> 4) << 3);
#pragma unroll
    for (int r = 0; r < 8; ++r) {
        float* row = G + (size_t)(rbase + r) * NG;
        row[n0 +  0 + col] = acc0[r];
        row[n0 + 16 + col] = acc1[r];
        row[n0 + 32 + col] = acc2[r];
        row[n0 + 48 + col] = acc3[r];
    }
}

// Elementwise cell update: gates order i,f,g,o
__global__ void lstm_update_kernel(const float* __restrict__ G, const float* __restrict__ bias,
                                   float* __restrict__ c_state, half_t* __restrict__ h_out) {
    int tid = blockIdx.x * 256 + threadIdx.x;     // 256*256
    int b = tid >> 8;
    int h = tid & 255;
    const float* g = G + (size_t)b * NG;
    float gi = g[h      ] + bias[h      ];
    float gf = g[h + 256] + bias[h + 256];
    float gg = g[h + 512] + bias[h + 512];
    float go = g[h + 768] + bias[h + 768];
    float c  = sigf(gf) * c_state[tid] + sigf(gi) * tanhf(gg);
    c_state[tid] = c;
    h_out[tid] = (half_t)(sigf(go) * tanhf(c));
}

// ---------------------------------------------------------------------------
// MLP head: feat = [h1 | s] (257) -> 256 relu -> 128 relu -> softplus
// grid 256 (one block per batch row), block 256
// ---------------------------------------------------------------------------
__global__ void __launch_bounds__(256) mlp_head_kernel(
        const half_t* __restrict__ h1, const float* __restrict__ s,
        const float* __restrict__ w1, const float* __restrict__ b1,
        const float* __restrict__ w2, const float* __restrict__ b2,
        const float* __restrict__ wp, const float* __restrict__ bp,
        float* __restrict__ out) {
    __shared__ float feat[257], z1[256], z2[128];
    const int b = blockIdx.x;
    const int t = threadIdx.x;
    feat[t] = (float)h1[b * HH + t];
    if (t == 0) feat[256] = s[b];
    __syncthreads();
    float a1 = b1[t];
    for (int k = 0; k < 257; ++k) a1 += feat[k] * w1[k * 256 + t];
    z1[t] = fmaxf(a1, 0.f);
    __syncthreads();
    if (t < 128) {
        float a2 = b2[t];
        for (int k = 0; k < 256; ++k) a2 += z1[k] * w2[k * 128 + t];
        z2[t] = fmaxf(a2, 0.f);
    }
    __syncthreads();
    if (t == 0) {
        float a = bp[0];
        for (int k = 0; k < 128; ++k) a += z2[k] * wp[k];
        out[b] = (a > 20.f) ? a : log1pf(__expf(a));
    }
}

// ---------------------------------------------------------------------------
// Host-side launch
// ---------------------------------------------------------------------------
extern "C" void kernel_launch(void* const* d_in, const int* in_sizes, int n_in,
                              void* d_out, int out_size, void* d_ws, size_t ws_size,
                              hipStream_t stream) {
    const float* x        = (const float*)d_in[0];
    const float* s        = (const float*)d_in[1];
    const float* conv1_w  = (const float*)d_in[2];
    const float* conv1_b  = (const float*)d_in[3];
    const float* bn1_g    = (const float*)d_in[4];
    const float* bn1_b    = (const float*)d_in[5];
    const float* conv2_w  = (const float*)d_in[6];
    const float* conv2_b  = (const float*)d_in[7];
    const float* bn2_g    = (const float*)d_in[8];
    const float* bn2_b    = (const float*)d_in[9];
    const float* lstm_wi0 = (const float*)d_in[10];
    const float* lstm_bi0 = (const float*)d_in[11];
    const float* lstm_wh0 = (const float*)d_in[12];
    const float* lstm_bh0 = (const float*)d_in[13];
    const float* lstm_wi1 = (const float*)d_in[14];
    const float* lstm_bi1 = (const float*)d_in[15];
    const float* lstm_wh1 = (const float*)d_in[16];
    const float* lstm_bh1 = (const float*)d_in[17];
    const float* w1       = (const float*)d_in[18];
    const float* b1       = (const float*)d_in[19];
    const float* w2       = (const float*)d_in[20];
    const float* b2       = (const float*)d_in[21];
    const float* wp       = (const float*)d_in[22];
    const float* bp       = (const float*)d_in[23];
    float* out = (float*)d_out;

    // ---- workspace layout (aliased big buffers) ----
    char* ws = (char*)d_ws;
    float*  y1  = (float*)(ws + 0);                     // 256*64*1024 f32 (67,108,864 B)
    float*  y2  = (float*)(ws + 0);                     // alias (y1 dead before conv2 writes)
    size_t  o   = 67108864;
    half_t* y1t = (half_t*)(ws + o);                    // 256*1024*64 f16 (33,554,432 B)
    half_t* X   = (half_t*)(ws + o);                    // alias (y1t dead before X written)
    o += 33554432;
    half_t* W0T = (half_t*)(ws + o); o += (size_t)NG * 384 * 2;   // 786,432
    half_t* W1T = (half_t*)(ws + o); o += (size_t)NG * 512 * 2;   // 1,048,576
    half_t* W2H = (half_t*)(ws + o); o += (size_t)5 * C2 * C1 * 2; o = (o + 255) & ~(size_t)255;
    float*  G   = (float*)(ws + o);  o += (size_t)BATCH * NG * 4; // 1,048,576
    // zero-init region: h0, h1 (f16), c0, c1 (f32)
    char*   zreg = ws + o;
    half_t* h0  = (half_t*)(ws + o); o += (size_t)BATCH * HH * 2;
    half_t* h1  = (half_t*)(ws + o); o += (size_t)BATCH * HH * 2;
    float*  c0  = (float*)(ws + o);  o += (size_t)BATCH * HH * 4;
    float*  c1  = (float*)(ws + o);  o += (size_t)BATCH * HH * 4;
    size_t  zreg_bytes = (size_t)BATCH * HH * (2 + 2 + 4 + 4);
    float*  bias0 = (float*)(ws + o); o += NG * 4;
    float*  bias1 = (float*)(ws + o); o += NG * 4;
    // stats zero region
    char*   streg = ws + o;
    float*  sum1 = (float*)(ws + o); o += C1 * 4;
    float*  sq1  = (float*)(ws + o); o += C1 * 4;
    float*  sum2 = (float*)(ws + o); o += C2 * 4;
    float*  sq2  = (float*)(ws + o); o += C2 * 4;
    size_t  streg_bytes = (C1 + C1 + C2 + C2) * 4;
    float*  scale1 = (float*)(ws + o); o += C1 * 4;
    float*  shift1 = (float*)(ws + o); o += C1 * 4;
    float*  scale2 = (float*)(ws + o); o += C2 * 4;
    float*  shift2 = (float*)(ws + o); o += C2 * 4;

    hipMemsetAsync(zreg, 0, zreg_bytes, stream);
    hipMemsetAsync(streg, 0, streg_bytes, stream);

    // ---- weight / bias prep ----
    prep_wcat_kernel<<<(NG * 384 + 255) / 256, 256, 0, stream>>>(lstm_wi0, C2, lstm_wh0, HH, W0T);
    prep_wcat_kernel<<<(NG * 512 + 255) / 256, 256, 0, stream>>>(lstm_wi1, HH, lstm_wh1, HH, W1T);
    prep_w2_kernel<<<(5 * C2 * C1 + 255) / 256, 256, 0, stream>>>(conv2_w, W2H);
    prep_bias_kernel<<<(NG + 255) / 256, 256, 0, stream>>>(lstm_bi0, lstm_bh0, lstm_bi1, lstm_bh1,
                                                           bias0, bias1);

    // ---- conv stage ----
    conv1_kernel<<<BATCH * 16, 256, 0, stream>>>(x, conv1_w, conv1_b, y1, sum1, sq1);
    bn_prep_kernel<<<1, 128, 0, stream>>>(sum1, sq1, bn1_g, bn1_b, scale1, shift1,
                                          C1, 1.0f / (float)(BATCH * L1));
    y1_norm_t_kernel<<<(BATCH * L1 * C1) / 256, 256, 0, stream>>>(y1, scale1, shift1, y1t);
    conv2_wmma_kernel<<<BATCH * 16, 256, 0, stream>>>(y1t, W2H, conv2_b, y2, sum2, sq2);
    bn_prep_kernel<<<1, 128, 0, stream>>>(sum2, sq2, bn2_g, bn2_b, scale2, shift2,
                                          C2, 1.0f / (float)(BATCH * TT));
    y2_norm_t_kernel<<<(TT * BATCH * C2) / 256, 256, 0, stream>>>(y2, scale2, shift2, X);

    // ---- LSTM scan: per step, layer0 gates -> update -> layer1 gates -> update ----
    for (int t = 0; t < TT; ++t) {
        const half_t* Xt = X + (size_t)t * BATCH * C2;
        gates_gemm_kernel<<<32, 256, 0, stream>>>(Xt, C2, C2, h0, HH, C2 + HH, W0T, C2 + HH, G);
        lstm_update_kernel<<<(BATCH * HH) / 256, 256, 0, stream>>>(G, bias0, c0, h0);
        gates_gemm_kernel<<<32, 256, 0, stream>>>(h0, HH, HH, h1, HH, HH + HH, W1T, HH + HH, G);
        lstm_update_kernel<<<(BATCH * HH) / 256, 256, 0, stream>>>(G, bias1, c1, h1);
    }

    // ---- MLP head ----
    mlp_head_kernel<<<BATCH, 256, 0, stream>>>(h1, s, w1, b1, w2, b2, wp, bp, out);

    (void)in_sizes; (void)n_in; (void)out_size; (void)ws_size;
}
